// TSP_MCTS_Network_9088150798980
// MI455X (gfx1250) — matrix operand order repaired and checked by reading, not compiled, not run on hardware
//
#include <hip/hip_runtime.h>
#include <hip/hip_bf16.h>

#define BATCH 4096
#define VN    256
#define HN    64
#define SROW  (257 * 256)   // floats per batch element of `state`

typedef __attribute__((ext_vector_type(16))) __bf16 v16bf;
typedef __attribute__((ext_vector_type(8)))  float  v8f;

struct U32B { uint4 lo, hi; };  // 32-byte carrier for v16bf bit_cast

__device__ __forceinline__ __bf16 f2bf(float x) {
    __hip_bfloat16 h = __float2bfloat16(x);
    return __builtin_bit_cast(__bf16, h);
}
__device__ __forceinline__ float bf2f(__bf16 b) {
    return __bfloat162float(__builtin_bit_cast(__hip_bfloat16, b));
}

__device__ __forceinline__ v16bf lds_load16bf(const __bf16* p) {
    U32B u;
    u.lo = *reinterpret_cast<const uint4*>(p);
    u.hi = *reinterpret_cast<const uint4*>(p + 8);
    return __builtin_bit_cast(v16bf, u);
}

// ---------------------------------------------------------------------------
// Kernel 1: per-batch (256x256)x(256x64) GEMM in split-bf16 (bf16x3) WMMA,
// fused relu + visited*b_vis bias + dot(w2) epilogue -> q_values (B,V).
// 512 threads = 16 waves; wave w computes rows [16w, 16w+16) of batch b.
// ---------------------------------------------------------------------------
__global__ __launch_bounds__(512) void gemm_q_kernel(
    const float* __restrict__ state, const float* __restrict__ W1,
    const float* __restrict__ b_vis, const float* __restrict__ w2,
    float* __restrict__ qout)
{
    __shared__ __bf16 Bhi[HN * VN];   // W1 hi part, column-major: [n*256 + k]
    __shared__ __bf16 Blo[HN * VN];   // W1 lo part

    const int b   = blockIdx.x;
    const int tid = threadIdx.x;

    // Stage W1 split into hi/lo bf16 (column-major so B tiles are contiguous)
    for (int idx = tid; idx < VN * HN; idx += 512) {
        const int k = idx >> 6;        // 0..255
        const int n = idx & 63;        // 0..63
        const float w   = W1[idx];     // W1[k][n]
        const __bf16 hi = f2bf(w);
        Bhi[n * VN + k] = hi;
        Blo[n * VN + k] = f2bf(w - bf2f(hi));
    }
    __syncthreads();

    const int wave = tid >> 5;
    const int lane = tid & 31;
    const int grp  = lane >> 4;        // 0: K-low half, 1: K-high half
    const int lof  = lane & 15;
    const int rowBase = wave * 16;

    // A row this lane feeds (A layout: lanes 0-15 and 16-31 both hold M=lane&15)
    const float* arow =
        state + (size_t)b * SROW + (size_t)(1 + rowBase + lof) * VN;

    // per-lane epilogue constants: n = t*16 + lof
    float bvv[4], wvv[4];
#pragma unroll
    for (int t = 0; t < 4; ++t) {
        bvv[t] = b_vis[t * 16 + lof];
        wvv[t] = w2[t * 16 + lof];
    }

    v8f acc[4] = {{}, {}, {}, {}};

    for (int kk = 0; kk < VN; kk += 32) {
        // ---- load A chunk (16 f32 per lane), split into hi/lo bf16 ----
        const float* ap = arow + kk + grp * 16;
        if (kk + 32 < VN) __builtin_prefetch(ap + 32, 0, 3);
        const float4 x0 = *reinterpret_cast<const float4*>(ap);
        const float4 x1 = *reinterpret_cast<const float4*>(ap + 4);
        const float4 x2 = *reinterpret_cast<const float4*>(ap + 8);
        const float4 x3 = *reinterpret_cast<const float4*>(ap + 12);
        float xs[16] = {x0.x, x0.y, x0.z, x0.w, x1.x, x1.y, x1.z, x1.w,
                        x2.x, x2.y, x2.z, x2.w, x3.x, x3.y, x3.z, x3.w};
        v16bf ah, al;
#pragma unroll
        for (int i = 0; i < 16; ++i) {
            const __bf16 h = f2bf(xs[i]);
            ah[i] = h;
            al[i] = f2bf(xs[i] - bf2f(h));
        }

        // ---- 4 N-tiles of 16 columns; 3 split-bf16 WMMAs each ----
#pragma unroll
        for (int t = 0; t < 4; ++t) {
            const int n = t * 16 + lof;
            const __bf16* bp = &Bhi[n * VN + kk + grp * 16];
            const __bf16* bq = &Blo[n * VN + kk + grp * 16];
            const v16bf bh = lds_load16bf(bp);
            const v16bf bl = lds_load16bf(bq);
            acc[t] = __builtin_amdgcn_wmma_f32_16x16x32_bf16(
                false, ah, false, bh, (short)0, acc[t], false, false);
            acc[t] = __builtin_amdgcn_wmma_f32_16x16x32_bf16(
                false, ah, false, bl, (short)0, acc[t], false, false);
            acc[t] = __builtin_amdgcn_wmma_f32_16x16x32_bf16(
                false, al, false, bh, (short)0, acc[t], false, false);
        }
    }

    // ---- fused epilogue: relu(acc + visited*b_vis) . w2 -> q[b, m] ----
    // C/D layout: VGPR r holds (M = r + grp*8, N = t*16 + lof)
#pragma unroll
    for (int r = 0; r < 8; ++r) {
        const int m = rowBase + r + grp * 8;
        const float vis = state[(size_t)b * SROW + m];   // visited[b, m]
        float qp = 0.0f;
#pragma unroll
        for (int t = 0; t < 4; ++t) {
            const float h = fmaxf(acc[t][r] + vis * bvv[t], 0.0f);
            qp += h * wvv[t];
        }
        // reduce across the 16 lanes sharing this m (xor 8,4,2,1 stays in-half)
        qp += __shfl_xor(qp, 8);
        qp += __shfl_xor(qp, 4);
        qp += __shfl_xor(qp, 2);
        qp += __shfl_xor(qp, 1);
        if (lof == 0) qout[b * VN + m] = qp;
    }
}

// ---------------------------------------------------------------------------
// Kernel 2: softmax stats over axis 0 (per column v: max_b, sum_b exp)
// ---------------------------------------------------------------------------
__global__ __launch_bounds__(256) void col_stats_kernel(
    const float* __restrict__ q, float* __restrict__ colmax,
    float* __restrict__ colsum)
{
    const int v   = blockIdx.x;
    const int tid = threadIdx.x;
    __shared__ float sred[256];

    float vals[16];
    float m = -3.4e38f;
#pragma unroll
    for (int i = 0; i < 16; ++i) {
        vals[i] = q[(size_t)(i * 256 + tid) * VN + v];
        m = fmaxf(m, vals[i]);
    }
    sred[tid] = m;
    __syncthreads();
    for (int s = 128; s > 0; s >>= 1) {
        if (tid < s) sred[tid] = fmaxf(sred[tid], sred[tid + s]);
        __syncthreads();
    }
    const float gmax = sred[0];
    __syncthreads();

    float ssum = 0.0f;
#pragma unroll
    for (int i = 0; i < 16; ++i) ssum += expf(vals[i] - gmax);
    sred[tid] = ssum;
    __syncthreads();
    for (int s = 128; s > 0; s >>= 1) {
        if (tid < s) sred[tid] += sred[tid + s];
        __syncthreads();
    }
    if (tid == 0) {
        colmax[v] = gmax;
        colsum[v] = sred[0];
    }
}

// ---------------------------------------------------------------------------
// Kernel 3: per-batch policy normalize + value head
// ---------------------------------------------------------------------------
__global__ __launch_bounds__(256) void finalize_kernel(
    const float* __restrict__ q, const float* __restrict__ state,
    const float* __restrict__ Wv, const float* __restrict__ bv,
    const float* __restrict__ colmax, const float* __restrict__ colsum,
    float* __restrict__ policy, float* __restrict__ value)
{
    const int b = blockIdx.x;
    const int v = threadIdx.x;
    __shared__ float sred[256];

    const float qv   = q[(size_t)b * VN + v];
    const float soft = expf(qv - colmax[v]) / colsum[v];
    const float vis  = state[(size_t)b * SROW + v];           // visited[b, v]
    const float temp = (soft + 0.01f) * (vis == 0.0f ? 1.0f : 0.0f);

    sred[v] = temp;
    __syncthreads();
    for (int s = 128; s > 0; s >>= 1) {
        if (v < s) sred[v] += sred[v + s];
        __syncthreads();
    }
    const float rowsum = sred[0];
    __syncthreads();

    const float pol = (rowsum == 0.0f) ? ((v == 0) ? 1.0f : 0.0f)
                                       : temp / rowsum;
    policy[(size_t)b * VN + v] = pol;

    // value = q[b,:] . Wv + bv
    sred[v] = qv * Wv[v];
    __syncthreads();
    for (int s = 128; s > 0; s >>= 1) {
        if (v < s) sred[v] += sred[v + s];
        __syncthreads();
    }
    if (v == 0) value[b] = sred[0] + bv[0];
}

// ---------------------------------------------------------------------------
extern "C" void kernel_launch(void* const* d_in, const int* in_sizes, int n_in,
                              void* d_out, int out_size, void* d_ws,
                              size_t ws_size, hipStream_t stream)
{
    const float* state = (const float*)d_in[0];   // (B, 257, 256)
    const float* W1    = (const float*)d_in[1];   // (256, 64)
    const float* b_vis = (const float*)d_in[2];   // (64,)
    const float* w2    = (const float*)d_in[3];   // (64,)
    const float* Wv    = (const float*)d_in[4];   // (256,)
    const float* bv    = (const float*)d_in[5];   // (1,)

    float* qout   = (float*)d_out;                         // (B, V)
    float* policy = qout + (size_t)BATCH * VN;             // (B, V)
    float* value  = qout + (size_t)2 * BATCH * VN;         // (B,)

    float* colmax = (float*)d_ws;                          // 256 floats
    float* colsum = colmax + VN;                           // 256 floats

    gemm_q_kernel<<<BATCH, 512, 0, stream>>>(state, W1, b_vis, w2, qout);
    col_stats_kernel<<<VN, 256, 0, stream>>>(qout, colmax, colsum);
    finalize_kernel<<<BATCH, 256, 0, stream>>>(qout, state, Wv, bv, colmax,
                                               colsum, policy, value);
}
